// Attention_24781961298297
// MI455X (gfx1250) — matrix operand order repaired and checked by reading, not compiled, not run on hardware
//
#include <hip/hip_runtime.h>
#include <hip/hip_bf16.h>

typedef _Float16 v16h __attribute__((ext_vector_type(16)));
typedef float    v8f  __attribute__((ext_vector_type(8)));

#define NB 8
#define SLE 4096
#define SLD 4096
#define DE 512
#define DA 128
#define MROWS 64
#define EPADF 516  // f32 row pad: +4 bank shift/row -> near-conflict-free b128 frag reads

// ---------------- kernel 1: u = Pu @ pv ; zero ctx ----------------
__global__ __launch_bounds__(128) void prep_small(const float* __restrict__ Pu,
                                                  const float* __restrict__ pv,
                                                  float* __restrict__ u,
                                                  float* __restrict__ ctx) {
  int a = threadIdx.x;
  float s = 0.f;
  for (int c = 0; c < DA; ++c) s += Pu[a * DA + c] * pv[c];
  u[a] = s;
  for (int i = a; i < NB * DA; i += DA) ctx[i] = 0.f;
}

// ------- kernel 2: transpose + f32->f16 convert Wk / Wv (col-major) -------
__global__ __launch_bounds__(128) void prep_w(const float* __restrict__ Wk,
                                              const float* __restrict__ Wv,
                                              _Float16* __restrict__ wkt,
                                              _Float16* __restrict__ wvt) {
  int g = blockIdx.x >> 7;
  int col = blockIdx.x & 127;
  const float* W = g ? Wv : Wk;
  _Float16* Wt = g ? wvt : wkt;
  for (int k = threadIdx.x; k < DE; k += 128)
    Wt[col * DE + k] = (_Float16)W[(size_t)k * DA + col];
}

// ---------------- kernel 3: fused K/V projection via WMMA ----------------
// grid = NB * (SLE/MROWS) blocks of 256 threads (8 waves).
// E tile staged f32 via async global->LDS DMA (ASYNCcnt), converted to f16
// while building A fragments. waves 0-3: K projection (-> k_term), 4-7: V.
__global__ __launch_bounds__(256) void fused_kv_gemm(const float* __restrict__ E,
                                                     const _Float16* __restrict__ wkt,
                                                     const _Float16* __restrict__ wvt,
                                                     const float* __restrict__ bk,
                                                     const float* __restrict__ bv,
                                                     const float* __restrict__ u,
                                                     float* __restrict__ Vout,
                                                     float* __restrict__ KTout) {
  __shared__ float sEf[MROWS][EPADF];   // 64 x 516 f32 = 129 KB
  __shared__ float sKT[MROWS];

  const int tid = threadIdx.x;
  const int b = blockIdx.x >> 6;       // 64 row-tiles per batch
  const int jbase = (blockIdx.x & 63) * MROWS;

  // ---- stage E tile (64 x 512 f32) into LDS with async DMA (no VGPR trip) ----
  const float* Eb = E + ((size_t)(b * SLE + jbase)) * DE;
  const unsigned int ldsBase = (unsigned int)(uintptr_t)&sEf[0][0];
  #pragma unroll
  for (int i = 0; i < 32; ++i) {
    int idx16 = i * 256 + tid;         // 16-byte chunk index; 32 chunks per row
    int row = idx16 >> 5;
    int c = (idx16 & 31) << 2;         // f32 column
    unsigned int ldsoff = ldsBase + (unsigned int)(row * EPADF + c) * 4u;
    unsigned int goff = (unsigned int)(row * DE + c) * 4u;
    asm volatile("global_load_async_to_lds_b128 %0, %1, %2 offset:0"
                 :: "v"(ldsoff), "v"(goff), "s"(Eb) : "memory");
  }
  if (tid < MROWS) sKT[tid] = 0.0f;
  asm volatile("s_wait_asynccnt 0x0" ::: "memory");
  __syncthreads();

  const int lane = tid & 31;
  const int wid = tid >> 5;
  const int g = wid >> 2;              // 0 = K projection, 1 = V projection
  const int nb = (wid & 3) << 5;       // this wave's 32-column slab
  const int n0 = nb + (lane & 15);
  const int n1 = n0 + 16;
  const int hi = lane >> 4;            // half-wave select

  const _Float16* Wt = g ? wvt : wkt;
  const float* bias = g ? bv : bk;
  const float b0v = bias[n0];
  const float b1v = bias[n1];

  v8f acc[4][2];
  #pragma unroll
  for (int mt = 0; mt < 4; ++mt)
    #pragma unroll
    for (int r = 0; r < 8; ++r) { acc[mt][0][r] = b0v; acc[mt][1][r] = b1v; }

  // B fragment (32x16 f16): lane = col n; lanes 0-15 hold K k..k+15, 16-31 K k+16..k+31
  const _Float16* wc0 = Wt + n0 * DE + hi * 16;
  const _Float16* wc1 = Wt + n1 * DE + hi * 16;
  // A fragment (16x32 f16): lane&15 = row; lanes 0-15 hold K k..k+15, 16-31 K k+16..k+31
  const int arow = lane & 15;
  const int aoff = hi * 16;

  #pragma unroll
  for (int ks = 0; ks < 16; ++ks) {
    const int kb = ks * 32;
    v16h bf0 = *(const v16h*)(wc0 + kb);
    v16h bf1 = *(const v16h*)(wc1 + kb);
    #pragma unroll
    for (int mt = 0; mt < 4; ++mt) {
      const float* ap = &sEf[mt * 16 + arow][kb + aoff];
      float4 q0 = *(const float4*)(ap);
      float4 q1 = *(const float4*)(ap + 4);
      float4 q2 = *(const float4*)(ap + 8);
      float4 q3 = *(const float4*)(ap + 12);
      v16h af;
      af[0]  = (_Float16)q0.x; af[1]  = (_Float16)q0.y;
      af[2]  = (_Float16)q0.z; af[3]  = (_Float16)q0.w;
      af[4]  = (_Float16)q1.x; af[5]  = (_Float16)q1.y;
      af[6]  = (_Float16)q1.z; af[7]  = (_Float16)q1.w;
      af[8]  = (_Float16)q2.x; af[9]  = (_Float16)q2.y;
      af[10] = (_Float16)q2.z; af[11] = (_Float16)q2.w;
      af[12] = (_Float16)q3.x; af[13] = (_Float16)q3.y;
      af[14] = (_Float16)q3.z; af[15] = (_Float16)q3.w;
      acc[mt][0] = __builtin_amdgcn_wmma_f32_16x16x32_f16(
          false, af, false, bf0, (short)0, acc[mt][0], false, false);
      acc[mt][1] = __builtin_amdgcn_wmma_f32_16x16x32_f16(
          false, af, false, bf1, (short)0, acc[mt][1], false, false);
    }
  }

  // relu
  #pragma unroll
  for (int mt = 0; mt < 4; ++mt)
    #pragma unroll
    for (int t = 0; t < 2; ++t)
      #pragma unroll
      for (int r = 0; r < 8; ++r)
        acc[mt][t][r] = fmaxf(acc[mt][t][r], 0.0f);

  if (g == 1) {
    // V projection: C/D layout -> VGPR r holds rows r (lanes 0-15) / r+8 (lanes 16-31)
    float* vb = Vout + ((size_t)(b * SLE + jbase)) * DA;
    #pragma unroll
    for (int mt = 0; mt < 4; ++mt)
      #pragma unroll
      for (int r = 0; r < 8; ++r) {
        int row = mt * 16 + hi * 8 + r;
        vb[row * DA + n0] = acc[mt][0][r];
        vb[row * DA + n1] = acc[mt][1][r];
      }
  } else {
    // K projection: reduce k_term[row] = sum_n K[row][n]*u[n]
    const float u0 = u[n0], u1 = u[n1];
    #pragma unroll
    for (int mt = 0; mt < 4; ++mt)
      #pragma unroll
      for (int r = 0; r < 8; ++r) {
        float s = acc[mt][0][r] * u0 + acc[mt][1][r] * u1;
        s += __shfl_xor(s, 1, 32);
        s += __shfl_xor(s, 2, 32);
        s += __shfl_xor(s, 4, 32);
        s += __shfl_xor(s, 8, 32);
        if ((lane & 15) == 0)
          atomicAdd(&sKT[mt * 16 + hi * 8 + r], s);   // ds_add_f32, 4 K-waves
      }
  }
  __syncthreads();
  if (tid < MROWS) KTout[b * SLE + jbase + tid] = sKT[tid];
}

// ---------------- kernel 4: per-batch softmax max & sum ----------------
__global__ __launch_bounds__(256) void softmax_stats(const float* __restrict__ kt,
                                                     float* __restrict__ stats) {
  __shared__ float red[256];
  const int b = blockIdx.x, tid = threadIdx.x;
  const float* k = kt + b * SLE;
  float m = -3.0e38f;
  for (int j = tid; j < SLE; j += 256) m = fmaxf(m, k[j]);
  red[tid] = m;
  __syncthreads();
  for (int s = 128; s > 0; s >>= 1) {
    if (tid < s) red[tid] = fmaxf(red[tid], red[tid + s]);
    __syncthreads();
  }
  m = red[0];
  __syncthreads();
  const float invs = 0.088388347648318447f;  // 1/sqrt(128)
  float z = 0.f;
  for (int j = tid; j < SLE; j += 256) z += __expf((k[j] - m) * invs);
  red[tid] = z;
  __syncthreads();
  for (int s = 128; s > 0; s >>= 1) {
    if (tid < s) red[tid] += red[tid + s];
    __syncthreads();
  }
  if (tid == 0) { stats[b] = m; stats[8 + b] = red[0]; }
}

// ------------- kernel 5: ctx[b,:] += sum_j w_j * v[b,j,:] -------------
__global__ __launch_bounds__(128) void context_accum(const float* __restrict__ kt,
                                                     const float* __restrict__ stats,
                                                     const float* __restrict__ V,
                                                     float* __restrict__ ctx) {
  const int b = blockIdx.x >> 5;
  const int chunk = blockIdx.x & 31;
  const int d = threadIdx.x;
  const float m = stats[b];
  const float invs = 0.088388347648318447f;
  const float* kb = kt + b * SLE + chunk * 128;
  const float* vb = V + ((size_t)(b * SLE + chunk * 128)) * DA;
  float acc = 0.f;
  #pragma unroll 4
  for (int jj = 0; jj < 128; ++jj) {
    float w = __expf((kb[jj] - m) * invs);  // uniform per block -> scalarized
    acc += w * vb[jj * DA + d];
  }
  atomicAdd(&ctx[b * DA + d], acc);         // global_atomic_add_f32
}

// ------------- kernel 6: out[b,i,:] = ctx[b,:]/Z[b] for all i -------------
__global__ __launch_bounds__(256) void broadcast_out(const float* __restrict__ ctx,
                                                     const float* __restrict__ stats,
                                                     float* __restrict__ out) {
  int idx = blockIdx.x * 1024 + threadIdx.x * 4;   // element index, multiple of 4
  int b = idx >> 19;                               // SLD*DA = 2^19 per batch
  int d = idx & (DA - 1);
  float zin = 1.0f / stats[8 + b];
  const float* c = ctx + b * DA + d;
  float4 o;
  o.x = c[0] * zin; o.y = c[1] * zin; o.z = c[2] * zin; o.w = c[3] * zin;
  *(float4*)(out + idx) = o;
}

extern "C" void kernel_launch(void* const* d_in, const int* in_sizes, int n_in,
                              void* d_out, int out_size, void* d_ws, size_t ws_size,
                              hipStream_t stream) {
  (void)in_sizes; (void)n_in; (void)out_size; (void)ws_size;
  const float* enc = (const float*)d_in[0];
  // d_in[1] (decoder), d_in[2] (Wq), d_in[3] (bq), d_in[8] (Pw): dead — q_term
  // cancels inside the softmax, so weights/output are independent of q.
  const float* Wk = (const float*)d_in[4];
  const float* bk = (const float*)d_in[5];
  const float* Wv = (const float*)d_in[6];
  const float* bv = (const float*)d_in[7];
  const float* Pu = (const float*)d_in[9];
  const float* pv = (const float*)d_in[10];
  float* out = (float*)d_out;

  char* ws = (char*)d_ws;
  float*    wsV   = (float*)(ws + 0);          // 8*4096*128 f32 = 16 MiB
  float*    wsKT  = (float*)(ws + 16777216);   // 8*4096 f32
  float*    wsU   = (float*)(ws + 16908288);   // 128 f32
  float*    wsSt  = (float*)(ws + 16908800);   // m[8], Z[8]
  float*    wsCtx = (float*)(ws + 16909056);   // 8*128 f32
  _Float16* wkt   = (_Float16*)(ws + 16913408); // 512*128 f16
  _Float16* wvt   = (_Float16*)(ws + 17044480); // 512*128 f16

  prep_small<<<1, 128, 0, stream>>>(Pu, pv, wsU, wsCtx);
  prep_w<<<256, 128, 0, stream>>>(Wk, Wv, wkt, wvt);
  fused_kv_gemm<<<NB * (SLE / MROWS), 256, 0, stream>>>(enc, wkt, wvt, bk, bv,
                                                        wsU, wsV, wsKT);
  softmax_stats<<<NB, 256, 0, stream>>>(wsKT, wsSt);
  context_accum<<<NB * 32, 128, 0, stream>>>(wsKT, wsSt, wsV, wsCtx);
  broadcast_out<<<1024, 256, 0, stream>>>(wsCtx, wsSt, out);
}
